// ExpertChoiceRouter_81844896792932
// MI455X (gfx1250) — matrix operand order repaired and checked by reading, not compiled, not run on hardware
//
#include <hip/hip_runtime.h>
#include <hip/hip_bf16.h>
#include <math.h>

// ---------------------------------------------------------------------------
// Problem constants (from reference setup_inputs)
// ---------------------------------------------------------------------------
#define NTOK 8192      // B*S = 4*2048
#define DIM  1024      // D
#define HID  4096      // 4*D
#define NEXP 8         // E
#define CAPQ 1280      // ceil(1.25*N/E)

typedef _Float16 v16h __attribute__((ext_vector_type(16)));
typedef _Float16 v8h  __attribute__((ext_vector_type(8)));
typedef _Float16 v4h  __attribute__((ext_vector_type(4)));
typedef float    v8f  __attribute__((ext_vector_type(8)));

// Async global->LDS copy path (ASYNCcnt-tracked), guarded per-toolchain.
#if __has_builtin(__builtin_amdgcn_global_load_async_to_lds_b128) && \
    __has_builtin(__builtin_amdgcn_s_wait_asynccnt)
#define HAVE_ASYNC_LDS 1
typedef int v4i_vs __attribute__((__vector_size__(16)));
typedef __attribute__((address_space(1))) v4i_vs as1_v4i;
typedef __attribute__((address_space(3))) v4i_vs as3_v4i;
#else
#define HAVE_ASYNC_LDS 0
#endif

// ---------------------------------------------------------------------------
// Kernel: grid-stride float4 zero fill
// ---------------------------------------------------------------------------
__global__ void zero_f4_kernel(float4* __restrict__ p, long long n4) {
    long long i = (long long)blockIdx.x * blockDim.x + threadIdx.x;
    long long stride = (long long)gridDim.x * blockDim.x;
    float4 z; z.x = 0.f; z.y = 0.f; z.z = 0.f; z.w = 0.f;
    for (; i < n4; i += stride) p[i] = z;
}

// ---------------------------------------------------------------------------
// Kernel: f32 -> f16 elementwise convert (x), float4 in / 4xf16 out
// ---------------------------------------------------------------------------
__global__ void convert_x_kernel(const float4* __restrict__ src,
                                 v4h* __restrict__ dst, long long n4) {
    long long i = (long long)blockIdx.x * blockDim.x + threadIdx.x;
    long long stride = (long long)gridDim.x * blockDim.x;
    for (; i < n4; i += stride) {
        float4 f = src[i];
        v4h h;
        h[0] = (_Float16)f.x; h[1] = (_Float16)f.y;
        h[2] = (_Float16)f.z; h[3] = (_Float16)f.w;
        dst[i] = h;
    }
}

// ---------------------------------------------------------------------------
// Kernel: W1 [DIM][HID] f32 -> W1^T [HID][DIM] f16, LDS-tiled 64x64 transpose
//   grid = (DIM/64, HID/64), block = 256
// ---------------------------------------------------------------------------
__global__ __launch_bounds__(256) void transpose_w1_kernel(
    const float* __restrict__ w1g, _Float16* __restrict__ w1t16) {
    __shared__ _Float16 tile[64 * 72];
    const int kBase = blockIdx.x * 64;
    const int nBase = blockIdx.y * 64;
    for (int i = threadIdx.x; i < 64 * 64; i += 256) {
        int k = i >> 6, n = i & 63;   // coalesced over n
        tile[k * 72 + n] = (_Float16)w1g[(size_t)(kBase + k) * HID + nBase + n];
    }
    __syncthreads();
    for (int i = threadIdx.x; i < 64 * 64; i += 256) {
        int n = i >> 6, k = i & 63;   // coalesced over k
        w1t16[(size_t)(nBase + n) * DIM + kBase + k] = tile[k * 72 + n];
    }
}

// ---------------------------------------------------------------------------
// Kernel: fused router MLP  scores = relu(x@W1 + b1) @ W2 + b2
//   grid = (64, 8); block = 256 (8 waves); wave w computes tokens [w*16,+16)
//   First layer via v_wmma_f32_16x16x32_f16; panels staged into LDS with
//   global_load_async_to_lds_b128 when available; 2nd layer fused through LDS.
// ---------------------------------------------------------------------------
#define XP_LD 72    // LDS row stride (halves) for 64-wide k panels (+8 pad)
#define H_LD  136   // LDS row stride (halves) for 128-wide h tile (+8 pad)

__global__ __launch_bounds__(256) void router_gemm_kernel(
    const _Float16* __restrict__ x16,    // [NTOK, DIM]
    const _Float16* __restrict__ w1t16,  // [HID, DIM] (transposed)
    const float* __restrict__ b1g,       // [HID]
    const float* __restrict__ w2g,       // [HID, NEXP]
    const float* __restrict__ b2g,       // [NEXP]
    float* __restrict__ scores)          // [NTOK, NEXP]  (pre-zeroed)
{
    // 36 KB aliased region: {x panel 18KB | W1^T panel 18KB} OR {h tile 34KB}
    __shared__ __align__(16) unsigned char s_raw[2 * 128 * XP_LD * 2];
    __shared__ float s_w2[128 * NEXP];                       // 4 KB
    _Float16* s_x  = (_Float16*)s_raw;                       // [128][XP_LD]
    _Float16* s_w1 = (_Float16*)(s_raw + 128 * XP_LD * 2);   // [128][XP_LD]
    _Float16* s_h  = (_Float16*)s_raw;                       // [128][H_LD]

    const int tid  = threadIdx.x;
    const int wave = tid >> 5;
    const int lane = tid & 31;
    const int tokBase = blockIdx.x * 128;
    const int yBase   = blockIdx.y * 512;

    // A-fragment addressing (16x32 f16): lane&15 = row M; lanes>=16 take K+8/K+24
    const int aRow  = (wave << 4) + (lane & 15);
    const int koffA = (lane >> 4) << 3;    // 0 or 8
    // B-fragment addressing (32x16 f16): lane&15 = col N; lanes>=16 take K+8/K+24
    const int koffB = (lane >> 4) << 3;    // 0 or 8
    const int nCol  = lane & 15;
    // C/D layout: element v -> (m = v + 8*(lane>>4), n = lane&15)
    const int mHi   = (lane >> 4) << 3;

#if HAVE_ASYNC_LDS
    const int rowInWave = lane >> 3;         // 0..3
    const int byteInRow = (lane & 7) * 16;   // 0..112 (16B units)
#endif

    // Per-thread score accumulators for the fused 2nd layer:
    float sacc[4];
#pragma unroll
    for (int p = 0; p < 4; ++p) {
        int idx = tid + (p << 8);
        sacc[p] = (blockIdx.y == 0) ? b2g[idx & 7] : 0.0f;
    }

    for (int chunk = 0; chunk < 4; ++chunk) {
        const int nChunkBase = yBase + chunk * 128;

        v8f acc[8];
#pragma unroll
        for (int t = 0; t < 8; ++t)
#pragma unroll
            for (int v = 0; v < 8; ++v) acc[t][v] = 0.0f;

        for (int kp = 0; kp < DIM / 64; ++kp) {
            const int kBase = kp * 64;
            __syncthreads();  // guard previous panel / previous-chunk h reads

#if HAVE_ASYNC_LDS
            // Async stage both panels: each b128 op moves 4 rows x 128B.
            // 8 ops per wave cover 16 rows of x panel + 16 rows of W1^T panel.
#pragma unroll
            for (int q = 0; q < 4; ++q) {
                int row = (wave << 4) + (q << 2) + rowInWave;   // 0..127
                const char* gx = (const char*)(x16 +
                    (size_t)(tokBase + row) * DIM + kBase) + byteInRow;
                char* lx = (char*)(s_x + row * XP_LD) + byteInRow;
                __builtin_amdgcn_global_load_async_to_lds_b128(
                    (as1_v4i*)gx, (as3_v4i*)lx, 0, 0);
                const char* gw = (const char*)(w1t16 +
                    (size_t)(nChunkBase + row) * DIM + kBase) + byteInRow;
                char* lw = (char*)(s_w1 + row * XP_LD) + byteInRow;
                __builtin_amdgcn_global_load_async_to_lds_b128(
                    (as1_v4i*)gw, (as3_v4i*)lw, 0, 0);
            }
            __builtin_amdgcn_s_wait_asynccnt(0);
#else
            // Fallback: vectorized synchronous staging (16B per thread)
            for (int i = tid; i < 128 * 8; i += 256) {
                int row = i >> 3, c = (i & 7) << 3;
                *(v8h*)&s_x[row * XP_LD + c] =
                    *(const v8h*)&x16[(size_t)(tokBase + row) * DIM + kBase + c];
                *(v8h*)&s_w1[row * XP_LD + c] =
                    *(const v8h*)&w1t16[(size_t)(nChunkBase + row) * DIM + kBase + c];
            }
            if (kBase + 64 < DIM)
                __builtin_prefetch(
                    &w1t16[(size_t)(nChunkBase + (tid >> 1)) * DIM + kBase + 64], 0, 1);
#endif
            __syncthreads();

#pragma unroll
            for (int ks = 0; ks < 2; ++ks) {
                const _Float16* ab = &s_x[aRow * XP_LD + ks * 32];
                v8h alo = *(const v8h*)(ab + koffA);
                v8h ahi = *(const v8h*)(ab + 16 + koffA);
                v16h A;
#pragma unroll
                for (int j = 0; j < 8; ++j) { A[j] = alo[j]; A[8 + j] = ahi[j]; }
#pragma unroll
                for (int t = 0; t < 8; ++t) {
                    const _Float16* bb =
                        &s_w1[(t * 16 + nCol) * XP_LD + ks * 32 + koffB];
                    v8h b0 = *(const v8h*)bb;
                    v8h b1 = *(const v8h*)(bb + 16);
                    v16h B;
#pragma unroll
                    for (int j = 0; j < 8; ++j) { B[j] = b0[j]; B[8 + j] = b1[j]; }
                    acc[t] = __builtin_amdgcn_wmma_f32_16x16x32_f16(
                        false, A, false, B, (short)0, acc[t], false, false);
                }
            }
        }
        __syncthreads();  // all fragment reads from panels complete

        // bias + ReLU, spill h tile (f16) to LDS (aliases panel memory)
#pragma unroll
        for (int t = 0; t < 8; ++t) {
            int nLoc = t * 16 + nCol;
            float bias = b1g[nChunkBase + nLoc];
#pragma unroll
            for (int v = 0; v < 8; ++v) {
                float hv = acc[t][v] + bias;
                hv = hv > 0.0f ? hv : 0.0f;
                s_h[((wave << 4) + mHi + v) * H_LD + nLoc] = (_Float16)hv;
            }
        }
        for (int i = tid; i < 128 * NEXP; i += 256)
            s_w2[i] = w2g[(size_t)nChunkBase * NEXP + i];
        __syncthreads();

        // fused second layer: scores[tok][e] += sum_n h[tok][n] * W2[n][e]
#pragma unroll
        for (int p = 0; p < 4; ++p) {
            int idx = tid + (p << 8);
            int tok = idx >> 3, e = idx & 7;
            float s = 0.0f;
#pragma unroll 8
            for (int n = 0; n < 128; ++n)
                s += (float)s_h[tok * H_LD + n] * s_w2[n * NEXP + e];
            sacc[p] += s;
        }
    }

    // commit partial scores (one atomic per (token, expert) per block)
#pragma unroll
    for (int p = 0; p < 4; ++p) {
        int idx = tid + (p << 8);
        int tok = idx >> 3, e = idx & 7;
        atomicAdd(&scores[(size_t)(tokBase + tok) * NEXP + e], sacc[p]);
    }
}

// ---------------------------------------------------------------------------
// Kernel: per-expert top-CAPQ via in-LDS bitonic sort of 8192 (score, idx)
//   order: descending score, ascending index on ties (matches lax.top_k)
// ---------------------------------------------------------------------------
__global__ __launch_bounds__(1024) void topk_kernel(
    const float* __restrict__ scores,     // [NTOK, NEXP]
    float* __restrict__ out_top_scores,   // [NEXP, CAPQ]
    float* __restrict__ out_top_idx)      // [NEXP, CAPQ] (as float)
{
    __shared__ float key[NTOK];
    __shared__ int   kid[NTOK];
    const int e = blockIdx.x;
    const int tid = threadIdx.x;

    for (int t = tid; t < NTOK; t += 1024) {
        key[t] = scores[(size_t)t * NEXP + e];
        kid[t] = t;
    }
    __syncthreads();

    for (int k = 2; k <= NTOK; k <<= 1) {
        for (int j = k >> 1; j > 0; j >>= 1) {
#pragma unroll
            for (int p = 0; p < 4; ++p) {
                int i = tid + (p << 10);
                int ixj = i ^ j;
                if (ixj > i) {
                    float ka = key[i], kb = key[ixj];
                    int   ia = kid[i], ib = kid[ixj];
                    bool aFirst = (ka > kb) || (ka == kb && ia < ib);
                    bool desc = ((i & k) == 0);
                    if (desc != aFirst) {
                        key[i] = kb; key[ixj] = ka;
                        kid[i] = ib; kid[ixj] = ia;
                    }
                }
            }
            __syncthreads();
        }
    }

    for (int c = tid; c < CAPQ; c += 1024) {
        out_top_scores[e * CAPQ + c] = key[c];
        out_top_idx[e * CAPQ + c]    = (float)kid[c];
    }
}

// ---------------------------------------------------------------------------
// Kernel: scatter ones into (pre-zeroed) assignment from top_indices
// ---------------------------------------------------------------------------
__global__ void scatter_ones_kernel(const float* __restrict__ tidx,
                                    float* __restrict__ assign) {
    int row = blockIdx.x * blockDim.x + threadIdx.x;
    if (row < NEXP * CAPQ) {
        int tok = (int)tidx[row];
        assign[(size_t)row * NTOK + tok] = 1.0f;
    }
}

// ---------------------------------------------------------------------------
// Kernel: gather dispatched tokens  out[row] = x[idx[row]]  (float4 copies)
// ---------------------------------------------------------------------------
__global__ __launch_bounds__(256) void gather_kernel(
    const float* __restrict__ xg, const float* __restrict__ tidx,
    float* __restrict__ out_disp)
{
    int row = blockIdx.x;
    int tok = (int)tidx[row];
    const float4* src = (const float4*)(xg + (size_t)tok * DIM);
    float4* dst = (float4*)(out_disp + (size_t)row * DIM);
    dst[threadIdx.x] = src[threadIdx.x];
}

// ---------------------------------------------------------------------------
// Kernel: aux losses. importance = softmax(scores).sum(0); load = CAPQ each.
// ---------------------------------------------------------------------------
__global__ __launch_bounds__(256) void loss_kernel(
    const float* __restrict__ scores, float* __restrict__ out_ll,
    float* __restrict__ out_il)
{
    __shared__ float red[256 * NEXP];
    float loc[NEXP];
#pragma unroll
    for (int e = 0; e < NEXP; ++e) loc[e] = 0.0f;

    for (int t = threadIdx.x; t < NTOK; t += 256) {
        float r[NEXP];
        float mx = -1e30f;
#pragma unroll
        for (int e = 0; e < NEXP; ++e) {
            r[e] = scores[(size_t)t * NEXP + e];
            mx = r[e] > mx ? r[e] : mx;
        }
        float sum = 0.0f;
#pragma unroll
        for (int e = 0; e < NEXP; ++e) { r[e] = __expf(r[e] - mx); sum += r[e]; }
        float inv = 1.0f / sum;
#pragma unroll
        for (int e = 0; e < NEXP; ++e) loc[e] += r[e] * inv;
    }
#pragma unroll
    for (int e = 0; e < NEXP; ++e) red[threadIdx.x * NEXP + e] = loc[e];
    __syncthreads();
    for (int s = 128; s > 0; s >>= 1) {
        if (threadIdx.x < s)
#pragma unroll
            for (int e = 0; e < NEXP; ++e)
                red[threadIdx.x * NEXP + e] += red[(threadIdx.x + s) * NEXP + e];
        __syncthreads();
    }
    if (threadIdx.x == 0) {
        float mean = 0.0f;
#pragma unroll
        for (int e = 0; e < NEXP; ++e) mean += red[e];
        mean *= (1.0f / NEXP);
        float var = 0.0f;
#pragma unroll
        for (int e = 0; e < NEXP; ++e) {
            float d = red[e] - mean;
            var += d * d;
        }
        var *= (1.0f / (NEXP - 1));
        *out_il = var / (mean * mean + 1e-6f);
        // every expert takes exactly CAPQ distinct tokens -> load variance = 0
        float lmean = (float)CAPQ + 1e-6f;
        *out_ll = 0.0f / (lmean * lmean + 1e-6f);
    }
}

// ---------------------------------------------------------------------------
// Host launcher.  Scratch (f16 copies + scores) lives inside the 335 MB
// assignment output region, which is zeroed AFTER the kernels that consume
// the scratch -> no dependence on ws_size at all.
// ---------------------------------------------------------------------------
extern "C" void kernel_launch(void* const* d_in, const int* in_sizes, int n_in,
                              void* d_out, int out_size, void* d_ws, size_t ws_size,
                              hipStream_t stream) {
    (void)in_sizes; (void)n_in; (void)out_size; (void)d_ws; (void)ws_size;

    const float* xg  = (const float*)d_in[0];
    const float* w1g = (const float*)d_in[1];
    const float* b1g = (const float*)d_in[2];
    const float* w2g = (const float*)d_in[3];
    const float* b2g = (const float*)d_in[4];

    float* out = (float*)d_out;
    float* out_disp   = out;                                    // [E, CAP, D]
    float* out_assign = out_disp + (size_t)NEXP * CAPQ * DIM;   // [E, CAP, N]
    float* out_tscore = out_assign + (size_t)NEXP * CAPQ * NTOK;// [E, CAP]
    float* out_tidx   = out_tscore + (size_t)NEXP * CAPQ;       // [E, CAP]
    float* out_ll     = out_tidx + (size_t)NEXP * CAPQ;
    float* out_il     = out_ll + 1;

    // Scratch carved from the assignment region (24.25 MB of 335 MB):
    _Float16* x16   = (_Float16*)out_assign;              // [NTOK, DIM]  16 MB
    _Float16* w1t16 = x16 + (size_t)NTOK * DIM;           // [HID, DIM]    8 MB
    float*    scores = (float*)(w1t16 + (size_t)HID * DIM); // [NTOK, E] 256 KB

    // 1) f32 -> f16 conversions (x; W1 transposed)
    hipLaunchKernelGGL(convert_x_kernel, dim3(2048), dim3(256), 0, stream,
                       (const float4*)xg, (v4h*)x16,
                       (long long)NTOK * DIM / 4);
    hipLaunchKernelGGL(transpose_w1_kernel, dim3(DIM / 64, HID / 64),
                       dim3(256), 0, stream, w1g, w1t16);

    // 2) zero the scores accumulator
    hipLaunchKernelGGL(zero_f4_kernel, dim3(64), dim3(256), 0, stream,
                       (float4*)scores, (long long)(NTOK * NEXP / 4));

    // 3) fused router MLP -> scores (WMMA + async LDS staging)
    hipLaunchKernelGGL(router_gemm_kernel, dim3(NTOK / 128, HID / 512),
                       dim3(256), 0, stream, x16, w1t16, b1g, w2g, b2g, scores);

    // 4) aux losses (reads scores while still live)
    hipLaunchKernelGGL(loss_kernel, dim3(1), dim3(256), 0, stream,
                       scores, out_ll, out_il);

    // 5) per-expert top-k (bitonic) -> top_scores / top_indices
    hipLaunchKernelGGL(topk_kernel, dim3(NEXP), dim3(1024), 0, stream,
                       scores, out_tscore, out_tidx);

    // 6) zero the full assignment region (erases scratch)
    hipLaunchKernelGGL(zero_f4_kernel, dim3(4096), dim3(256), 0, stream,
                       (float4*)out_assign,
                       (long long)NEXP * CAPQ * NTOK / 4);

    // 7) scatter the 10240 ones
    hipLaunchKernelGGL(scatter_ones_kernel, dim3((NEXP * CAPQ + 255) / 256),
                       dim3(256), 0, stream, out_tidx, out_assign);

    // 8) gather dispatched token features (fp32 fidelity from original x)
    hipLaunchKernelGGL(gather_kernel, dim3(NEXP * CAPQ), dim3(256), 0, stream,
                       xg, out_tidx, out_disp);
}
